// LGPExtractor_1640677507535
// MI455X (gfx1250) — matrix operand order repaired and checked by reading, not compiled, not run on hardware
//
#include <hip/hip_runtime.h>
#include <math.h>

// ---------------------------------------------------------------------------
// Problem constants (from reference setup_inputs)
// ---------------------------------------------------------------------------
#define BB   8
#define CC   256
#define MM   2048
#define NN   4096
#define C3   768          // 3*C
#define HH   384          // hidden
#define OUTD 128

typedef __attribute__((ext_vector_type(16))) __bf16 v16bf;
typedef __attribute__((ext_vector_type(8)))  float  v8f;

// ---------------------------------------------------------------------------
// Workspace layout (bytes, 256-aligned)
// ---------------------------------------------------------------------------
#define OFF_FEATC   ((size_t)0)                          // (B,M,768) bf16 : 25165824
#define OFF_REFS    ((size_t)25165824)                   // (B,M,3)   f32  : 196608
#define OFF_IDX     ((size_t)25362432)                   // (B,N,3)   i32  : 393216
#define OFF_WGT     ((size_t)25755648)                   // (B,N,3)   f32  : 393216
#define OFF_XB      ((size_t)26148864)                   // (B,N,768) bf16 : 50331648
#define OFF_HT      ((size_t)76480512)                   // (B,N,384) bf16 : 25165824
#define OFF_W1BF    ((size_t)101646336)                  // (384,768) bf16 : 589824
#define OFF_W2BF    ((size_t)102236160)                  // (128,384) bf16 : 98304

// ---------------------------------------------------------------------------
// Kernel 1: rotate features into canonical frame, write bf16 (B, M, 768)
//   featc[b, m, c*3+j] = sum_i vn_feat[b,c,m,i] * R[b,i,j]
// ---------------------------------------------------------------------------
__global__ __launch_bounds__(256)
void rotate_feat_kernel(const float* __restrict__ vn_feat,
                        const float* __restrict__ R,
                        __bf16* __restrict__ featc) {
  size_t gid = (size_t)blockIdx.x * blockDim.x + threadIdx.x;   // (b*C + c)*M + m
  if (gid >= (size_t)BB * CC * MM) return;
  int m = (int)(gid % MM);
  int c = (int)((gid / MM) % CC);
  int b = (int)(gid / ((size_t)MM * CC));
  const float* f  = vn_feat + gid * 3;
  const float* Rb = R + b * 9;
  float x = f[0], y = f[1], z = f[2];
  __bf16* o = featc + ((size_t)b * MM + m) * C3 + c * 3;
  o[0] = (__bf16)(x * Rb[0] + y * Rb[3] + z * Rb[6]);
  o[1] = (__bf16)(x * Rb[1] + y * Rb[4] + z * Rb[7]);
  o[2] = (__bf16)(x * Rb[2] + y * Rb[5] + z * Rb[8]);
}

// ---------------------------------------------------------------------------
// Kernel 2: rotate reference xyz, write f32 (B, M, 3)
// ---------------------------------------------------------------------------
__global__ __launch_bounds__(256)
void rotate_xyz_kernel(const float* __restrict__ vn_xyz,
                       const float* __restrict__ R,
                       float* __restrict__ refs) {
  int gid = blockIdx.x * blockDim.x + threadIdx.x;  // b*M + m
  if (gid >= BB * MM) return;
  int m = gid % MM;
  int b = gid / MM;
  const float* Rb = R + b * 9;
  float x = vn_xyz[((size_t)b * 3 + 0) * MM + m];
  float y = vn_xyz[((size_t)b * 3 + 1) * MM + m];
  float z = vn_xyz[((size_t)b * 3 + 2) * MM + m];
  float* o = refs + (size_t)gid * 3;
  o[0] = x * Rb[0] + y * Rb[3] + z * Rb[6];
  o[1] = x * Rb[1] + y * Rb[4] + z * Rb[7];
  o[2] = x * Rb[2] + y * Rb[5] + z * Rb[8];
}

// ---------------------------------------------------------------------------
// Kernel 3: f32 -> bf16 weight conversion (grid-stride)
// ---------------------------------------------------------------------------
__global__ __launch_bounds__(256)
void cvt_bf16_kernel(const float* __restrict__ src, __bf16* __restrict__ dst, int n) {
  for (int i = blockIdx.x * blockDim.x + threadIdx.x; i < n;
       i += gridDim.x * blockDim.x)
    dst[i] = (__bf16)src[i];
}

// ---------------------------------------------------------------------------
// Kernel 4: brute-force 3-NN + inverse-distance weights.
// Refs for one batch (2048*3 f32 = 24KB) live in LDS; 256 queries per block.
// ---------------------------------------------------------------------------
__global__ __launch_bounds__(256)
void knn_kernel(const float* __restrict__ refs,
                const float* __restrict__ target_xyz,
                int* __restrict__ idx, float* __restrict__ wgt) {
  __shared__ float sref[MM * 3];
  int b     = blockIdx.x >> 4;      // grid.x = B * (N/256) = 8*16
  int ntile = blockIdx.x & 15;
  const float* rb = refs + (size_t)b * MM * 3;
  for (int i = threadIdx.x; i < MM * 3; i += 256) sref[i] = rb[i];
  __syncthreads();

  int n = ntile * 256 + threadIdx.x;
  float qx = target_xyz[((size_t)b * 3 + 0) * NN + n];
  float qy = target_xyz[((size_t)b * 3 + 1) * NN + n];
  float qz = target_xyz[((size_t)b * 3 + 2) * NN + n];

  float d0 = 3.4e38f, d1 = 3.4e38f, d2 = 3.4e38f;
  int   i0 = 0, i1 = 0, i2 = 0;
  for (int m = 0; m < MM; ++m) {
    float dx = sref[m * 3 + 0] - qx;
    float dy = sref[m * 3 + 1] - qy;
    float dz = sref[m * 3 + 2] - qz;
    float d = dx * dx + dy * dy + dz * dz;
    if (d < d0)      { d2 = d1; i2 = i1; d1 = d0; i1 = i0; d0 = d; i0 = m; }
    else if (d < d1) { d2 = d1; i2 = i1; d1 = d;  i1 = m; }
    else if (d < d2) { d2 = d;  i2 = m; }
  }
  float w0 = 1.0f / (sqrtf(d0) + 1e-8f);
  float w1 = 1.0f / (sqrtf(d1) + 1e-8f);
  float w2 = 1.0f / (sqrtf(d2) + 1e-8f);
  float wsum = w0 + w1 + w2;
  size_t o = ((size_t)b * NN + n) * 3;
  idx[o + 0] = i0; idx[o + 1] = i1; idx[o + 2] = i2;
  wgt[o + 0] = w0 / wsum; wgt[o + 1] = w1 / wsum; wgt[o + 2] = w2 / wsum;
}

// ---------------------------------------------------------------------------
// Kernel 5: inverse-distance interpolation -> Xb (B, N, 768) bf16
// One block per (b, n); 768 contiguous channels split across 256 threads.
// ---------------------------------------------------------------------------
__global__ __launch_bounds__(256)
void interp_kernel(const __bf16* __restrict__ featc,
                   const int* __restrict__ idx,
                   const float* __restrict__ wgt,
                   __bf16* __restrict__ Xb) {
  int bn = blockIdx.x;                       // 0..B*N-1
  int b  = bn >> 12;                         // N = 4096
  size_t o = (size_t)bn * 3;
  int i0 = idx[o + 0], i1 = idx[o + 1], i2 = idx[o + 2];
  float w0 = wgt[o + 0], w1 = wgt[o + 1], w2 = wgt[o + 2];
  const __bf16* r0 = featc + ((size_t)b * MM + i0) * C3;
  const __bf16* r1 = featc + ((size_t)b * MM + i1) * C3;
  const __bf16* r2 = featc + ((size_t)b * MM + i2) * C3;
  __bf16* op = Xb + (size_t)bn * C3;
  for (int c = threadIdx.x; c < C3; c += 256)
    op[c] = (__bf16)(w0 * (float)r0[c] + w1 * (float)r1[c] + w2 * (float)r2[c]);
}

// ---------------------------------------------------------------------------
// WMMA fragment loaders (contiguous k order, same for A and B, wave32)
//  A (16x32, 16-bit): lane L row = L&15; halves 0..15 -> K=kb..kb+15,
//                     kb = (L>>4)*16
//  B (32x16, 16-bit): lane L col = L&15; halves 0..15 -> K=kb..kb+15,
//                     kb = (L>>4)*16   (rows of the N-major activation array)
// ---------------------------------------------------------------------------
__device__ __forceinline__ v16bf load_a_frag(const __bf16* __restrict__ rowp,
                                             int k0, int lane) {
  const __bf16* p = rowp + k0 + ((lane >> 4) << 4);
  v16bf v;
#pragma unroll
  for (int i = 0; i < 8; ++i) v[i] = p[i];
#pragma unroll
  for (int i = 0; i < 8; ++i) v[i + 8] = p[i + 8];
  return v;
}

__device__ __forceinline__ v16bf load_b_frag(const __bf16* __restrict__ rowp,
                                             int k0, int lane) {
  const __bf16* p = rowp + k0 + ((lane >> 4) << 4);
  v16bf v;
#pragma unroll
  for (int i = 0; i < 16; ++i) v[i] = p[i];
  return v;
}

// ---------------------------------------------------------------------------
// Kernel 6: GEMM1  Ht(b,n,o) = relu( W1[o,:] . Xb[b,n,:] + b1[o] ), bf16 out
// 8 waves/block; each wave: 16 (M) x 64 (N) strip, K = 768 in steps of 32.
// grid = (N/512, H/16, B)
// ---------------------------------------------------------------------------
__global__ __launch_bounds__(256)
void gemm1_kernel(const __bf16* __restrict__ W1bf,
                  const __bf16* __restrict__ Xb,
                  const float* __restrict__ b1,
                  __bf16* __restrict__ Ht) {
  const int lane  = threadIdx.x & 31;
  const int wave  = threadIdx.x >> 5;
  const int b     = blockIdx.z;
  const int mt    = blockIdx.y;                       // 0..23
  const int nbase = blockIdx.x * 512 + wave * 64;
  const int col   = lane & 15;

  const __bf16* arowp = W1bf + (size_t)(mt * 16 + col) * C3;
  const __bf16* xbase = Xb + (size_t)b * NN * C3;

  v8f acc[4] = {};
  for (int k0 = 0; k0 < C3; k0 += 32) {
    v16bf a = load_a_frag(arowp, k0, lane);
    __builtin_prefetch(arowp + k0 + 32, 0, 1);
#pragma unroll
    for (int s = 0; s < 4; ++s) {
      const __bf16* browp = xbase + (size_t)(nbase + s * 16 + col) * C3;
      v16bf bf = load_b_frag(browp, k0, lane);
      acc[s] = __builtin_amdgcn_wmma_f32_16x16x32_bf16(
          false, a, false, bf, (short)0, acc[s], false, false);
    }
  }

  const int mloc = (lane >> 4) << 3;                  // 0 or 8
#pragma unroll
  for (int s = 0; s < 4; ++s) {
    int n = nbase + s * 16 + col;
    __bf16* op = Ht + ((size_t)b * NN + n) * HH + mt * 16 + mloc;
#pragma unroll
    for (int r = 0; r < 8; ++r) {
      float v = acc[s][r] + b1[mt * 16 + mloc + r];
      v = v > 0.0f ? v : 0.0f;
      op[r] = (__bf16)v;
    }
  }
}

// ---------------------------------------------------------------------------
// Kernel 7: GEMM2  out(b,o,n) = W2[o,:] . Ht[b,n,:] + b2[o], f32 out (B,128,N)
// grid = (N/512, OUT/16, B)
// ---------------------------------------------------------------------------
__global__ __launch_bounds__(256)
void gemm2_kernel(const __bf16* __restrict__ W2bf,
                  const __bf16* __restrict__ Ht,
                  const float* __restrict__ b2,
                  float* __restrict__ out) {
  const int lane  = threadIdx.x & 31;
  const int wave  = threadIdx.x >> 5;
  const int b     = blockIdx.z;
  const int mt    = blockIdx.y;                       // 0..7
  const int nbase = blockIdx.x * 512 + wave * 64;
  const int col   = lane & 15;

  const __bf16* arowp = W2bf + (size_t)(mt * 16 + col) * HH;
  const __bf16* hbase = Ht + (size_t)b * NN * HH;

  v8f acc[4] = {};
  for (int k0 = 0; k0 < HH; k0 += 32) {
    v16bf a = load_a_frag(arowp, k0, lane);
#pragma unroll
    for (int s = 0; s < 4; ++s) {
      const __bf16* browp = hbase + (size_t)(nbase + s * 16 + col) * HH;
      v16bf bf = load_b_frag(browp, k0, lane);
      acc[s] = __builtin_amdgcn_wmma_f32_16x16x32_bf16(
          false, a, false, bf, (short)0, acc[s], false, false);
    }
  }

  const int mloc = (lane >> 4) << 3;
#pragma unroll
  for (int s = 0; s < 4; ++s) {
    int n = nbase + s * 16 + col;
#pragma unroll
    for (int r = 0; r < 8; ++r) {
      int m = mt * 16 + mloc + r;
      out[((size_t)b * OUTD + m) * NN + n] = acc[s][r] + b2[m];
    }
  }
}

// ---------------------------------------------------------------------------
// Launcher
// ---------------------------------------------------------------------------
extern "C" void kernel_launch(void* const* d_in, const int* in_sizes, int n_in,
                              void* d_out, int out_size, void* d_ws, size_t ws_size,
                              hipStream_t stream) {
  const float* vn_feat    = (const float*)d_in[0];
  const float* vn_xyz     = (const float*)d_in[1];
  const float* target_xyz = (const float*)d_in[2];
  const float* R_align    = (const float*)d_in[3];
  const float* W1         = (const float*)d_in[4];
  const float* b1         = (const float*)d_in[5];
  const float* W2         = (const float*)d_in[6];
  const float* b2         = (const float*)d_in[7];
  float* out = (float*)d_out;

  char* ws = (char*)d_ws;
  __bf16* featc = (__bf16*)(ws + OFF_FEATC);
  float*  refs  = (float*)(ws + OFF_REFS);
  int*    idx   = (int*)(ws + OFF_IDX);
  float*  wgt   = (float*)(ws + OFF_WGT);
  __bf16* Xb    = (__bf16*)(ws + OFF_XB);
  __bf16* Ht    = (__bf16*)(ws + OFF_HT);
  __bf16* W1bf  = (__bf16*)(ws + OFF_W1BF);
  __bf16* W2bf  = (__bf16*)(ws + OFF_W2BF);

  // 1) rotate features -> bf16 (B, M, 768)
  {
    size_t total = (size_t)BB * CC * MM;
    rotate_feat_kernel<<<(unsigned)((total + 255) / 256), 256, 0, stream>>>(
        vn_feat, R_align, featc);
  }
  // 2) rotate reference xyz -> f32 (B, M, 3)
  rotate_xyz_kernel<<<(BB * MM + 255) / 256, 256, 0, stream>>>(
      vn_xyz, R_align, refs);
  // 3) weights to bf16
  cvt_bf16_kernel<<<(HH * C3 + 255) / 256, 256, 0, stream>>>(W1, W1bf, HH * C3);
  cvt_bf16_kernel<<<(OUTD * HH + 255) / 256, 256, 0, stream>>>(W2, W2bf, OUTD * HH);
  // 4) 3-NN + inverse-distance weights
  knn_kernel<<<BB * (NN / 256), 256, 0, stream>>>(refs, target_xyz, idx, wgt);
  // 5) interpolation -> Xb (B, N, 768) bf16
  interp_kernel<<<BB * NN, 256, 0, stream>>>(featc, idx, wgt, Xb);
  // 6) GEMM1 (+bias+ReLU) -> Ht (B, N, 384) bf16
  gemm1_kernel<<<dim3(NN / 512, HH / 16, BB), 256, 0, stream>>>(W1bf, Xb, b1, Ht);
  // 7) GEMM2 (+bias) -> out (B, 128, N) f32
  gemm2_kernel<<<dim3(NN / 512, OUTD / 16, BB), 256, 0, stream>>>(W2bf, Ht, b2, out);
}
